// MambaBlock_64871186039086
// MI455X (gfx1250) — matrix operand order repaired and checked by reading, not compiled, not run on hardware
//
#include <hip/hip_runtime.h>

// ---------------------------------------------------------------------------
// Mamba block for MI455X (gfx1250, wave32).  GEMMs via v_wmma_f32_16x16x32_bf16
// with 16x64 register blocking (4 accumulators share one A fragment).
// Shapes: B=4, L=2048, R=1024, D=128, N=64, K=3.  BL = B*L = 8192.
// ---------------------------------------------------------------------------

typedef __attribute__((ext_vector_type(16))) __bf16 v16bf;
typedef __attribute__((ext_vector_type(8)))  __bf16 v8bf;
typedef __attribute__((ext_vector_type(8)))  float  v8f;
typedef __attribute__((ext_vector_type(4)))  float  v4f;

#define BL_   8192
#define LSEQ_ 2048
#define RDIM_ 1024
#define DDIM_ 128
#define NSTATE_ 64

// ---- WMMA operand loaders (16-bit layouts per CDNA5 ISA 7.12.2) ------------
// A-matrix 16x32 bf16: lane l (0-15) holds row M=l, K-chunks [0..7] and [16..23];
// lanes 16-31 hold K-chunks [8..15] and [24..31].  Two 16-byte loads.
__device__ __forceinline__ v16bf load_a_bf16(const __bf16* p) {
  union { v16bf v; struct { v8bf lo, hi; } s; } u;
  u.s.lo = *(const v8bf*)(p);
  u.s.hi = *(const v8bf*)(p + 16);
  return u.v;
}
// B-matrix 32x16 bf16 (stored as W^T, N-major rows of length K): lane l holds
// column N=l&15; lanes 0-15 K=0..7,16..23, lanes 16-31 K=8..15,24..31 (two 16-byte loads).
__device__ __forceinline__ v16bf load_b_bf16(const __bf16* p) {
  union { v16bf v; struct { v8bf lo, hi; } s; } u;
  u.s.lo = *(const v8bf*)(p);
  u.s.hi = *(const v8bf*)(p + 16);
  return u.v;
}

// ---- GEMM: C[M,Nc] = A[M,Kc](bf16) * Bt[Nc,Kc]^T + bias --------------------
// One wave computes a 16x64 output tile: 4 WMMA accumulators reuse one A frag.
__global__ __launch_bounds__(128) void gemm_bf16_4n(
    const __bf16* __restrict__ A, const __bf16* __restrict__ Bt,
    const float* __restrict__ bias, float* __restrict__ C,
    int M, int Nc, int Kc) {
  const int wave = blockIdx.x * (blockDim.x >> 5) + (threadIdx.x >> 5);
  const int lane = threadIdx.x & 31;
  const int ntx = Nc >> 6;                       // 64-wide n tiles
  if (wave >= (M >> 4) * ntx) return;            // wave-uniform: EXEC stays full
  const int mt = wave / ntx, nt = wave % ntx;
  const int arow = (mt << 4) + (lane & 15);
  const int ka = (lane >> 4) << 3;               // 0 or 8  (A chunk select)
  const int kb = (lane >> 4) << 3;               // 0 or 8  (B chunk select)

  const __bf16* ap  = A  + (size_t)arow * Kc + ka;
  const __bf16* bp0 = Bt + (size_t)((nt << 6) +  0 + (lane & 15)) * Kc + kb;
  const __bf16* bp1 = Bt + (size_t)((nt << 6) + 16 + (lane & 15)) * Kc + kb;
  const __bf16* bp2 = Bt + (size_t)((nt << 6) + 32 + (lane & 15)) * Kc + kb;
  const __bf16* bp3 = Bt + (size_t)((nt << 6) + 48 + (lane & 15)) * Kc + kb;

  v8f acc[4] = {};
  for (int k0 = 0; k0 < Kc; k0 += 32) {
    __builtin_prefetch(ap + k0 + 256, 0, 1);     // global_prefetch_b8 (A stream)
    const v16bf a = load_a_bf16(ap + k0);
    acc[0] = __builtin_amdgcn_wmma_f32_16x16x32_bf16(false, a, false,
                 load_b_bf16(bp0 + k0), (short)0, acc[0], false, false);
    acc[1] = __builtin_amdgcn_wmma_f32_16x16x32_bf16(false, a, false,
                 load_b_bf16(bp1 + k0), (short)0, acc[1], false, false);
    acc[2] = __builtin_amdgcn_wmma_f32_16x16x32_bf16(false, a, false,
                 load_b_bf16(bp2 + k0), (short)0, acc[2], false, false);
    acc[3] = __builtin_amdgcn_wmma_f32_16x16x32_bf16(false, a, false,
                 load_b_bf16(bp3 + k0), (short)0, acc[3], false, false);
  }
  // C/D layout: lanes 0-15 -> N=lane, M=m0+r; lanes 16-31 -> N=lane-16, M=m0+8+r
  const int m0 = (mt << 4) + ((lane >> 4) << 3);
  const int nlane = lane & 15;
#pragma unroll
  for (int i = 0; i < 4; ++i) {
    const int n = (nt << 6) + (i << 4) + nlane;
    const float bv = bias ? bias[n] : 0.0f;
#pragma unroll
    for (int r = 0; r < 8; ++r) C[(size_t)(m0 + r) * Nc + n] = acc[i][r] + bv;
  }
}

// ---- bulk f32 -> bf16 convert (vectorized x4) ------------------------------
__global__ __launch_bounds__(256) void cvt_bf16_x4(
    const float* __restrict__ in, __bf16* __restrict__ out, int n4) {
  int idx = blockIdx.x * blockDim.x + threadIdx.x;
  if (idx >= n4) return;
  const v4f v = *(const v4f*)(in + (size_t)idx * 4);
  __bf16* o = out + (size_t)idx * 4;
  o[0] = (__bf16)v[0]; o[1] = (__bf16)v[1];
  o[2] = (__bf16)v[2]; o[3] = (__bf16)v[3];
}

// ---- weight transpose + f32 -> bf16 (out[c*rows+r] = in[r*cols+c]) ---------
__global__ __launch_bounds__(256) void transpose_cvt(
    const float* __restrict__ in, __bf16* __restrict__ out, int rows, int cols) {
  int idx = blockIdx.x * blockDim.x + threadIdx.x;
  if (idx >= rows * cols) return;
  int r = idx / cols, c = idx - r * cols;
  out[(size_t)c * rows + r] = (__bf16)in[idx];
}

// ---- causal depthwise conv (K=3) + SiLU ------------------------------------
__global__ __launch_bounds__(256) void conv_silu(
    const float* __restrict__ u, const float* __restrict__ conv_w,
    const float* __restrict__ conv_b, float* __restrict__ xf,
    __bf16* __restrict__ xb) {
  int idx = blockIdx.x * blockDim.x + threadIdx.x;     // BL_ * DDIM_
  if (idx >= BL_ * DDIM_) return;
  const int d = idx & (DDIM_ - 1);
  const int rowbl = idx >> 7;
  const int l = rowbl & (LSEQ_ - 1);
  float s = conv_b[d] + conv_w[d * 3 + 2] * u[(size_t)rowbl * 256 + d];
  if (l >= 1) s += conv_w[d * 3 + 1] * u[(size_t)(rowbl - 1) * 256 + d];
  if (l >= 2) s += conv_w[d * 3 + 0] * u[(size_t)(rowbl - 2) * 256 + d];
  const float v = s / (1.0f + __expf(-s));             // SiLU
  xf[idx] = v;
  xb[idx] = (__bf16)v;
}

// ---- S6 selective scan ------------------------------------------------------
// Grid: 16 blocks = 4 batches x 4 d-chunks of 32.  256 threads/block.
// Thread owns (d = d0 + tid>>3, n in [ (tid&7)*8, +8 )) -> 8 register states.
__global__ __launch_bounds__(256) void scan_s6(
    const float* __restrict__ dbc,    // [BL,256] = {delta_pre | Bm_raw | Cm_raw}
    const float* __restrict__ xf,     // [BL,128]
    const float* __restrict__ u,      // [BL,256] (right gate in cols 128..255)
    const float* __restrict__ b_delta, const float* __restrict__ b_B,
    const float* __restrict__ b_C, const float* __restrict__ A_log,
    const float* __restrict__ D_skip, __bf16* __restrict__ yg) {
  const int b  = blockIdx.x >> 2;
  const int d0 = (blockIdx.x & 3) * 32;
  const int tid = threadIdx.x;
  const int d  = d0 + (tid >> 3);
  const int n0 = (tid & 7) * 8;

  float Av[8], h[8], bB[8], bC[8];
#pragma unroll
  for (int j = 0; j < 8; ++j) {
    Av[j] = -__expf(A_log[d * NSTATE_ + n0 + j]);
    h[j] = 0.0f;
    bB[j] = b_B[n0 + j];
    bC[j] = b_C[n0 + j];
  }
  const float bdel = b_delta[d];
  const float dsk  = D_skip[d];

  for (int t = 0; t < LSEQ_; ++t) {
    const size_t row = (size_t)b * LSEQ_ + t;
    const float dpre = dbc[row * 256 + d] + bdel;
    const float delta = (dpre > 20.0f) ? dpre : log1pf(__expf(dpre));  // softplus
    const float xv = xf[row * 128 + d];
    const float coef = delta * xv;
    const float* pb = dbc + row * 256 + 128 + n0;
    const float* pc = dbc + row * 256 + 192 + n0;
    float acc = 0.0f;
#pragma unroll
    for (int j = 0; j < 8; ++j) {
      const float Bn = pb[j] + bB[j];
      const float Cn = pc[j] + bC[j];
      const float dA = __expf(delta * Av[j]);
      h[j] = dA * h[j] + coef * Bn;
      acc += h[j] * Cn;
    }
    // reduce over the 8 lanes sharing d (aligned subgroup within the wave)
    acc += __shfl_xor(acc, 1, 8);
    acc += __shfl_xor(acc, 2, 8);
    acc += __shfl_xor(acc, 4, 8);
    if ((tid & 7) == 0) {
      const float y = acc + dsk * xv;
      const float r = u[row * 256 + 128 + d];
      const float g = y * (r / (1.0f + __expf(-r)));   // gate = y * silu(right)
      yg[row * 128 + d] = (__bf16)g;
    }
  }
}

// ---------------------------------------------------------------------------
extern "C" void kernel_launch(void* const* d_in, const int* in_sizes, int n_in,
                              void* d_out, int out_size, void* d_ws, size_t ws_size,
                              hipStream_t stream) {
  const float* up_x    = (const float*)d_in[0];
  const float* W_up    = (const float*)d_in[1];
  const float* b_up    = (const float*)d_in[2];
  const float* conv_w  = (const float*)d_in[3];
  const float* conv_b  = (const float*)d_in[4];
  const float* W_delta = (const float*)d_in[5];
  const float* b_delta = (const float*)d_in[6];
  const float* W_B     = (const float*)d_in[7];
  const float* b_B     = (const float*)d_in[8];
  const float* W_C     = (const float*)d_in[9];
  const float* b_C     = (const float*)d_in[10];
  const float* A_log   = (const float*)d_in[11];
  const float* D_skip  = (const float*)d_in[12];
  const float* W_down  = (const float*)d_in[13];
  const float* b_down  = (const float*)d_in[14];
  float* out = (float*)d_out;
  char* ws = (char*)d_ws;

  // 256-byte aligned workspace layout (~42 MB total)
  constexpr size_t OFF_U    = 0;                              // f32  [8192,256]
  constexpr size_t OFF_XF   = OFF_U    + 8192ull * 256 * 4;   // f32  [8192,128]
  constexpr size_t OFF_XB   = OFF_XF   + 8192ull * 128 * 4;   // bf16 [8192,128]
  constexpr size_t OFF_DBC  = OFF_XB   + 8192ull * 128 * 2;   // f32  [8192,256]
  constexpr size_t OFF_YG   = OFF_DBC  + 8192ull * 256 * 4;   // bf16 [8192,128]
  constexpr size_t OFF_WUP  = OFF_YG   + 8192ull * 128 * 2;   // bf16 [256,1024]
  constexpr size_t OFF_WCAT = OFF_WUP  + 256ull * 1024 * 2;   // bf16 [256,128]
  constexpr size_t OFF_WDN  = OFF_WCAT + 256ull * 128 * 2;    // bf16 [1024,128]
  constexpr size_t OFF_UPB  = OFF_WDN  + 1024ull * 128 * 2;   // bf16 [8192,1024]

  float*  u_buf = (float*)(ws + OFF_U);
  float*  xf    = (float*)(ws + OFF_XF);
  __bf16* xb    = (__bf16*)(ws + OFF_XB);
  float*  dbc   = (float*)(ws + OFF_DBC);
  __bf16* yg    = (__bf16*)(ws + OFF_YG);
  __bf16* WupT  = (__bf16*)(ws + OFF_WUP);
  __bf16* WcatT = (__bf16*)(ws + OFF_WCAT);
  __bf16* WdnT  = (__bf16*)(ws + OFF_WDN);
  __bf16* upxb  = (__bf16*)(ws + OFF_UPB);

  // 1) activation + weight conversion to bf16 (weights transposed to N-major)
  cvt_bf16_x4<<<(BL_ * RDIM_ / 4 + 255) / 256, 256, 0, stream>>>(up_x, upxb, BL_ * RDIM_ / 4);
  transpose_cvt<<<(1024 * 256 + 255) / 256, 256, 0, stream>>>(W_up, WupT, 1024, 256);
  transpose_cvt<<<(128 * 128 + 255) / 256, 256, 0, stream>>>(W_delta, WcatT, 128, 128);
  transpose_cvt<<<(128 * 64 + 255) / 256, 256, 0, stream>>>(W_B, WcatT + 128 * 128, 128, 64);
  transpose_cvt<<<(128 * 64 + 255) / 256, 256, 0, stream>>>(W_C, WcatT + 192 * 128, 128, 64);
  transpose_cvt<<<(128 * 1024 + 255) / 256, 256, 0, stream>>>(W_down, WdnT, 128, 1024);

  // 2) up-projection: u = up_x @ W_up + b_up        [8192,1024]x[1024,256]
  gemm_bf16_4n<<<512, 128, 0, stream>>>(upxb, WupT, b_up, u_buf, 8192, 256, 1024);

  // 3) causal depthwise conv + SiLU -> x (f32 + bf16)
  conv_silu<<<(BL_ * DDIM_ + 255) / 256, 256, 0, stream>>>(u_buf, conv_w, conv_b, xf, xb);

  // 4) fused delta/B/C projection: dbc = x @ [W_delta|W_B|W_C]   [8192,128]x[128,256]
  gemm_bf16_4n<<<512, 128, 0, stream>>>(xb, WcatT, nullptr, dbc, 8192, 256, 128);

  // 5) selective scan + skip + SiLU gate -> yg (bf16)
  scan_s6<<<16, 256, 0, stream>>>(dbc, xf, u_buf, b_delta, b_B, b_C, A_log, D_skip, yg);

  // 6) down-projection: out = yg @ W_down + b_down  [8192,128]x[128,1024]
  gemm_bf16_4n<<<2048, 128, 0, stream>>>(yg, WdnT, b_down, out, 8192, 1024, 128);
}
